// DSASingleHeadAttention_73967926771758
// MI455X (gfx1250) — matrix operand order repaired and checked by reading, not compiled, not run on hardware
//
#include <hip/hip_runtime.h>
#include <hip/hip_bf16.h>

// ---------------------------------------------------------------------------
// DSA single-head attention block for MI455X (gfx1250), wave32 + WMMA f16.
//   Shapes: B=8, C=256, D=C/2=128, H=W=64, N=4096, HID=1024.
// Pipeline:
//   1) prep:   dwconv3x3+BN+SiLU -> Q,K (f16, scale 1/16 folded into Q)
//              dwconv3x3+residual -> V^T (f16, [c][n] for contiguous B-loads)
//   2) wconv:  fold BN into p1, convert p1/p2/p3 -> f16
//   3) attn:   flash attention, v_wmma_f32_16x16x32_f16, online softmax,
//              K tile staged in LDS via global_load_async_to_lds_b128
//   4) proj:   z -> p1(+bias) -> p2+SiLU -> p3 -> *gamma + x (all WMMA)
// Workspace use: ~49.2 MB.
// ---------------------------------------------------------------------------

typedef _Float16 f16;
typedef __attribute__((ext_vector_type(16))) _Float16 v16h;
typedef __attribute__((ext_vector_type(8)))  _Float16 v8h;
typedef __attribute__((ext_vector_type(8)))  float    v8f;

#define BATCH 8
#define CDIM  256
#define DDIM  128
#define NPOS  4096
#define HID   1024
#define HW    64
#define KPAD  136   // padded K-tile row (128 + 8 halfs): bank-conflict-free B-loads

static __device__ inline v8f wmma16(v16h a, v16h b, v8f c) {
  // D = A(16x32 f16) * B(32x16 f16) + C(16x16 f32)
  return __builtin_amdgcn_wmma_f32_16x16x32_f16(
      false, a, false, b, (short)0, c, false, false);
}

// A-operand: per-lane row = lane%16; elems 0..7 = K kh..kh+7,
// elems 8..15 = K 16+kh..23+kh (kh = 8*(lane>=16)). p = row + 32s + kh.
static __device__ inline v16h load_a(const f16* p) {
  v8h lo = *(const v8h*)p;
  v8h up = *(const v8h*)(p + 16);
  v16h r;
#pragma unroll
  for (int j = 0; j < 8; ++j) { r[j] = lo[j]; r[j + 8] = up[j]; }
  return r;
}

// B-operand: per-lane col = lane%16; elems 0..7 = K at p, elems 8..15 = K at p+16 (16B-aligned).
static __device__ inline v16h load_b(const f16* p) {
  v8h lo = *(const v8h*)p;
  v8h up = *(const v8h*)(p + 16);
  v16h r;
#pragma unroll
  for (int j = 0; j < 8; ++j) { r[j] = lo[j]; r[j + 8] = up[j]; }
  return r;
}

static __device__ inline float rmax16(float v) {  // reduce within 16-lane half
  v = fmaxf(v, __shfl_xor(v, 1, 32));
  v = fmaxf(v, __shfl_xor(v, 2, 32));
  v = fmaxf(v, __shfl_xor(v, 4, 32));
  v = fmaxf(v, __shfl_xor(v, 8, 32));
  return v;
}
static __device__ inline float rsum16(float v) {
  v += __shfl_xor(v, 1, 32);
  v += __shfl_xor(v, 2, 32);
  v += __shfl_xor(v, 4, 32);
  v += __shfl_xor(v, 8, 32);
  return v;
}

// ---------------------------------------------------------------------------
// Kernel 1: fused depthwise convs + BN + SiLU, emit Q/K/V^T in f16.
// ---------------------------------------------------------------------------
__global__ __launch_bounds__(256) void prep_kernel(
    const float* __restrict__ x, const float* __restrict__ qkw,
    const float* __restrict__ qkg, const float* __restrict__ qkb,
    const float* __restrict__ qkm, const float* __restrict__ qkv,
    const float* __restrict__ pegw,
    f16* __restrict__ qh, f16* __restrict__ kh, f16* __restrict__ vT) {
  int idx = blockIdx.x * 256 + threadIdx.x;          // exact grid
  int w = idx & (HW - 1);
  int h = (idx >> 6) & (HW - 1);
  int c = (idx >> 12) & (CDIM - 1);
  int b = idx >> 20;
  const float* xc = x + ((size_t)(b * CDIM + c) << 12);
  float sqk = 0.f, speg = 0.f;
#pragma unroll
  for (int dh = -1; dh <= 1; ++dh) {
    int hh = h + dh;
    if (hh < 0 || hh >= HW) continue;
#pragma unroll
    for (int dw = -1; dw <= 1; ++dw) {
      int ww = w + dw;
      if (ww < 0 || ww >= HW) continue;
      float xv = xc[hh * HW + ww];
      int wi = c * 9 + 3 * (dh + 1) + (dw + 1);
      sqk  += xv * qkw[wi];
      speg += xv * pegw[wi];
    }
  }
  int n = (h << 6) | w;
  float sc = qkg[c] * rsqrtf(qkv[c] + 1e-5f);
  float y = sqk * sc + (qkb[c] - qkm[c] * sc);
  y = y / (1.f + __expf(-y));                        // SiLU
  if (c < DDIM)
    qh[((size_t)b * NPOS + n) * DDIM + c] = (f16)(y * 0.0625f);  // 1/sqrt(256)
  else
    kh[((size_t)b * NPOS + n) * DDIM + (c - DDIM)] = (f16)y;
  vT[((size_t)(b * CDIM + c) << 12) + n] = (f16)(speg + xc[n]);
}

// ---------------------------------------------------------------------------
// Kernel 2: fold BN into p1, convert weights to f16.
// ---------------------------------------------------------------------------
__global__ __launch_bounds__(256) void wconv_kernel(
    const float* __restrict__ p1, const float* __restrict__ pbg,
    const float* __restrict__ pbb, const float* __restrict__ pbm,
    const float* __restrict__ pbv, const float* __restrict__ p2,
    const float* __restrict__ p3,
    f16* __restrict__ w1, float* __restrict__ bias1,
    f16* __restrict__ w2, f16* __restrict__ w3) {
  int idx = blockIdx.x * 256 + threadIdx.x;
  if (idx < CDIM * CDIM) {
    int o = idx >> 8;
    float sc = pbg[o] * rsqrtf(pbv[o] + 1e-5f);
    w1[idx] = (f16)(p1[idx] * sc);
    if (idx < CDIM) {
      float s0 = pbg[idx] * rsqrtf(pbv[idx] + 1e-5f);
      bias1[idx] = pbb[idx] - pbm[idx] * s0;
    }
  } else if (idx < CDIM * CDIM + HID * CDIM) {
    int i = idx - CDIM * CDIM;
    w2[i] = (f16)p2[i];
  } else if (idx < CDIM * CDIM + 2 * HID * CDIM) {
    int i = idx - CDIM * CDIM - HID * CDIM;
    w3[i] = (f16)p3[i];
  }
}

// ---------------------------------------------------------------------------
// Kernel 3: flash attention.
// Block = 8 waves, one batch, one 64-query group. Wave = 16 query rows x
// 128 channels (waves pair per row-strip, redundant S, split channels).
// Per 64-key step: K tile (64x128 f16) staged in LDS once per block via
// CDNA5 async-to-LDS loads (ASYNCcnt); V^T streamed from global (L2).
// P transposed C-layout -> A-layout via per-wave LDS scratch.
// ---------------------------------------------------------------------------
__global__ __launch_bounds__(256) void attn_kernel(
    const f16* __restrict__ qh, const f16* __restrict__ kh,
    const f16* __restrict__ vT, f16* __restrict__ zh) {
  __shared__ __align__(64) f16 Klds[64][KPAD];   // 17.0 KB (padded)
  __shared__ __align__(64) f16 Plds[8][16][64];  // 16 KB, per-wave regions

  const int tid   = threadIdx.x;
  const int lane  = tid & 31;
  const int wv    = tid >> 5;
  const int b     = blockIdx.x >> 6;       // batch
  const int qg    = blockIdx.x & 63;       // 64-query group
  const int strip = wv & 3;
  const int chh   = wv >> 2;               // channel half
  const int qbase = qg * 64 + strip * 16;

  const int lrow = lane & 15;
  const int lh   = lane >> 4;
  const int kha  = lh * 8;    // A-operand K base
  const int khb  = lh * 8;    // B-operand K base

  // Resident Q tile: 16 rows x 128 d as 4 A-operands.
  v16h qa[4];
  {
    const f16* qr = qh + ((size_t)b * NPOS + qbase + lrow) * DDIM;
#pragma unroll
    for (int s = 0; s < 4; ++s) qa[s] = load_a(qr + 32 * s + kha);
  }

  v8f o[8];
#pragma unroll
  for (int t = 0; t < 8; ++t) o[t] = (v8f)(0.f);
  float mrow[8], lsum[8];
#pragma unroll
  for (int r = 0; r < 8; ++r) { mrow[r] = -1e30f; lsum[r] = 0.f; }

  const uint32_t kofs = (uint32_t)(uintptr_t)(&Klds[0][0]);  // LDS byte offset

  for (int mb = 0; mb < NPOS; mb += 64) {
    __syncthreads();  // previous iteration's consumers done with Klds
    // --- async-stage K tile: 1024 x 16B units, 4 per thread, coalesced ---
#pragma unroll
    for (int j = 0; j < 4; ++j) {
      int u = tid + 256 * j;              // 0..1023
      int key = u >> 4, dseg = u & 15;    // 16 units per 128-half row
      const f16* src =
          kh + ((size_t)b * NPOS + mb + key) * DDIM + dseg * 8;
      uint32_t dst = kofs + (uint32_t)(key * KPAD + dseg * 8) * 2;
      asm volatile("global_load_async_to_lds_b128 %0, %1, off"
                   :: "v"(dst), "v"(src) : "memory");
    }
    asm volatile("s_wait_asynccnt 0x0" ::: "memory");
    __syncthreads();  // whole tile visible to all waves

    // --- S = (Q/16) K^T : 4 col-tiles x (K=128 as 4x32) ---
    v8f sS[4];
#pragma unroll
    for (int ct = 0; ct < 4; ++ct) {
      v8f acc = (v8f)(0.f);
      const f16* kr = &Klds[16 * ct + lrow][khb];
#pragma unroll
      for (int s = 0; s < 4; ++s) acc = wmma16(qa[s], load_b(kr + 32 * s), acc);
      sS[ct] = acc;
    }
    // --- online softmax over these 64 keys ---
#pragma unroll
    for (int r = 0; r < 8; ++r) {
      float tm = fmaxf(fmaxf(sS[0][r], sS[1][r]), fmaxf(sS[2][r], sS[3][r]));
      float mnew = fmaxf(mrow[r], rmax16(tm));
      float alpha = __expf(mrow[r] - mnew);
      mrow[r] = mnew;
      float ps = 0.f;
#pragma unroll
      for (int ct = 0; ct < 4; ++ct) {
        float p = __expf(sS[ct][r] - mnew);
        sS[ct][r] = p;
        ps += p;
      }
      lsum[r] = lsum[r] * alpha + rsum16(ps);
#pragma unroll
      for (int t = 0; t < 8; ++t) o[t][r] *= alpha;
    }
    // --- transpose P (C-layout -> A-layout) through per-wave LDS ---
#pragma unroll
    for (int r = 0; r < 8; ++r) {
#pragma unroll
      for (int ct = 0; ct < 4; ++ct)
        Plds[wv][r + 8 * lh][16 * ct + lrow] = (f16)sS[ct][r];
    }
    v16h pa0 = load_a(&Plds[wv][lrow][kha]);
    v16h pa1 = load_a(&Plds[wv][lrow][32 + kha]);
    // --- O += P * V^T-slice (8 channel tiles, K=64 keys as 2x32) ---
#pragma unroll
    for (int t = 0; t < 8; ++t) {
      int c = 128 * chh + 16 * t + lrow;
      const f16* vr = vT + ((size_t)(b * CDIM + c) << 12) + mb + khb;
      o[t] = wmma16(pa0, load_b(vr), o[t]);
      o[t] = wmma16(pa1, load_b(vr + 32), o[t]);
    }
  }

  // --- normalize and store z rows [b*N+row][c] in f16 ---
#pragma unroll
  for (int r = 0; r < 8; ++r) {
    float inv = 1.f / lsum[r];
    int row = qbase + r + 8 * lh;
    f16* zr = zh + ((size_t)b * NPOS + row) * CDIM;
#pragma unroll
    for (int t = 0; t < 8; ++t)
      zr[128 * chh + 16 * t + lrow] = (f16)(o[t][r] * inv);
  }
}

// ---------------------------------------------------------------------------
// Kernel 4: projection MLP. Block = 64 positions, 8 waves.
// LDS (exactly 64 KB): Z[64][256] | H1[64][256]; H2 chunk aliases Z.
// ---------------------------------------------------------------------------
__global__ __launch_bounds__(256) void proj_kernel(
    const f16* __restrict__ zh, const f16* __restrict__ w1,
    const float* __restrict__ bias1, const f16* __restrict__ w2,
    const f16* __restrict__ w3, const float* __restrict__ gamma,
    const float* __restrict__ x, float* __restrict__ out) {
  __shared__ __align__(64) f16 smem[32768];  // 64 KB
  f16* Zt = smem;            // [64][256]
  f16* H1 = smem + 16384;    // [64][256]
  f16* H2 = smem;            // [64][128] (aliases dead Zt in phase 2)

  const int tid  = threadIdx.x;
  const int lane = tid & 31;
  const int wv   = tid >> 5;
  const int lrow = lane & 15;
  const int lh   = lane >> 4;
  const int kha  = lh * 8;
  const int khb  = lh * 8;
  const int rs   = wv & 3;        // row strip (16 rows)
  const int cg   = wv >> 2;       // channel group 0/1
  const int rowblk = blockIdx.x * 64;
  const int arow = 16 * rs + lrow;

  {  // stage Z tile (64x256 f16 = 32 KB), flat copy
    const uint4* src = (const uint4*)(zh + (size_t)rowblk * CDIM);
    uint4* dst = (uint4*)Zt;
    for (int i = tid; i < 2048; i += 256) dst[i] = src[i];
  }
  __syncthreads();

  // ---- phase 1: H1 = Z * W1^T + bias (BN folded) ----
#pragma unroll
  for (int t = 0; t < 8; ++t) {
    int ocol = 128 * cg + 16 * t + lrow;
    v8f acc = (v8f)(0.f);
#pragma unroll
    for (int s = 0; s < 8; ++s) {
      v16h a  = load_a(Zt + arow * CDIM + 32 * s + kha);
      v16h bm = load_b(w1 + (size_t)ocol * CDIM + 32 * s + khb);
      acc = wmma16(a, bm, acc);
    }
    float bb = bias1[ocol];
#pragma unroll
    for (int r = 0; r < 8; ++r)
      H1[(16 * rs + r + 8 * lh) * CDIM + ocol] = (f16)(acc[r] + bb);
  }
  __syncthreads();

  // ---- phase 2: out = W3 * silu(W2 * H1), hidden in 128-wide chunks ----
  v8f o3[8];
#pragma unroll
  for (int t = 0; t < 8; ++t) o3[t] = (v8f)(0.f);

  for (int hc = 0; hc < 8; ++hc) {
    int hbase = 128 * hc;
#pragma unroll
    for (int tt = 0; tt < 4; ++tt) {  // wave owns one 16-wide hidden column
      int hcol = 16 * wv + lrow;      // 0..127
      v8f acc = (v8f)(0.f);
#pragma unroll
      for (int s = 0; s < 8; ++s) {
        v16h a  = load_a(H1 + (16 * tt + lrow) * CDIM + 32 * s + kha);
        v16h bm = load_b(w2 + (size_t)(hbase + hcol) * CDIM + 32 * s + khb);
        acc = wmma16(a, bm, acc);
      }
#pragma unroll
      for (int r = 0; r < 8; ++r) {
        float y = acc[r];
        y = y / (1.f + __expf(-y));  // SiLU
        H2[(16 * tt + r + 8 * lh) * 128 + hcol] = (f16)y;
      }
    }
    __syncthreads();
#pragma unroll
    for (int t = 0; t < 8; ++t) {   // O3 += H2chunk * W3chunk^T
      int ocol = 128 * cg + 16 * t + lrow;
#pragma unroll
      for (int s = 0; s < 4; ++s) {
        v16h a  = load_a(H2 + arow * 128 + 32 * s + kha);
        v16h bm = load_b(w3 + (size_t)ocol * HID + hbase + 32 * s + khb);
        o3[t] = wmma16(a, bm, o3[t]);
      }
    }
    __syncthreads();
  }

  // ---- epilogue: out[b][c][n] = o3 * gamma[c] + x[b][c][n] ----
#pragma unroll
  for (int t = 0; t < 8; ++t) {
    int c = 128 * cg + 16 * t + lrow;
    float g = gamma[c];
#pragma unroll
    for (int r = 0; r < 8; ++r) {
      int grow = rowblk + 16 * rs + r + 8 * lh;
      int bb = grow >> 12, n = grow & (NPOS - 1);
      size_t xi = ((size_t)(bb * CDIM + c) << 12) + n;
      out[xi] = o3[t][r] * g + x[xi];
    }
  }
}

// ---------------------------------------------------------------------------
extern "C" void kernel_launch(void* const* d_in, const int* in_sizes, int n_in,
                              void* d_out, int out_size, void* d_ws, size_t ws_size,
                              hipStream_t stream) {
  (void)in_sizes; (void)n_in; (void)out_size; (void)ws_size;
  const float* x     = (const float*)d_in[0];
  const float* qkw   = (const float*)d_in[1];
  const float* qkg   = (const float*)d_in[2];
  const float* qkb   = (const float*)d_in[3];
  const float* qkm   = (const float*)d_in[4];
  const float* qkv   = (const float*)d_in[5];
  const float* pegw  = (const float*)d_in[6];
  const float* p1w   = (const float*)d_in[7];
  const float* pbg   = (const float*)d_in[8];
  const float* pbb   = (const float*)d_in[9];
  const float* pbm   = (const float*)d_in[10];
  const float* pbv   = (const float*)d_in[11];
  const float* p2w   = (const float*)d_in[12];
  const float* p3w   = (const float*)d_in[13];
  const float* gamma = (const float*)d_in[14];
  float* out = (float*)d_out;

  // workspace carve (all offsets 256B-aligned); total ~49.2 MB
  char* ws = (char*)d_ws;
  f16*   qh = (f16*)(ws + 0);           //  8 MB: [B][N][128]
  f16*   kh = (f16*)(ws + 8388608);     //  8 MB: [B][N][128]
  f16*   vT = (f16*)(ws + 16777216);    // 16 MB: [B][C][N]
  f16*   zh = (f16*)(ws + 33554432);    // 16 MB: [B*N][C]
  f16*   w1 = (f16*)(ws + 50331648);    // 128 KB
  float* b1 = (float*)(ws + 50462720);  //   1 KB
  f16*   w2 = (f16*)(ws + 50463744);    // 512 KB
  f16*   w3 = (f16*)(ws + 50988032);    // 512 KB

  prep_kernel<<<dim3(32768), dim3(256), 0, stream>>>(
      x, qkw, qkg, qkb, qkm, qkv, pegw, qh, kh, vT);
  wconv_kernel<<<dim3(2304), dim3(256), 0, stream>>>(
      p1w, pbg, pbb, pbm, pbv, p2w, p3w, w1, b1, w2, w3);
  attn_kernel<<<dim3(512), dim3(256), 0, stream>>>(qh, kh, vT, zh);
  proj_kernel<<<dim3(512), dim3(256), 0, stream>>>(
      zh, w1, b1, w2, w3, gamma, x, out);
}
